// VQEmbedding_52192442581295
// MI455X (gfx1250) — matrix operand order repaired and checked by reading, not compiled, not run on hardware
//
#include <hip/hip_runtime.h>
#include <hip/hip_bf16.h>

// Problem constants (match reference)
#define N_TOKENS 65536
#define DIM      256
#define K_CODES  1024
#define OUT_ZQ   (N_TOKENS * DIM)   // 16777216 floats of z_q_st
#define OUT_LOSS (OUT_ZQ)           // 1 float loss
#define OUT_IDX  (OUT_ZQ + 1)       // 65536 floats of indices

// Tiling
#define TOK_PER_BLK     128          // 8 waves x 16 rows
#define CODES_PER_CHUNK 64
#define NCHUNK          (K_CODES / CODES_PER_CHUNK)   // 16
#define GATHER_BLOCKS   16384        // (N_TOKENS*64 threads)/256

// LDS layout (bytes): A_hi 64KB | A_lo 64KB | B buf0 64KB | B buf1 64KB = 256KB
#define LDS_AHI   0
#define LDS_ALO   65536
#define LDS_B     131072
#define LDS_BSTEP 65536
#define LDS_BLO   32768   // lo half within a B buffer

typedef __attribute__((ext_vector_type(16))) __bf16 v16bf;
typedef __attribute__((ext_vector_type(8)))  __bf16 v8bf;
typedef __attribute__((ext_vector_type(8)))  float  v8f;

static __device__ __forceinline__ v8f wmma_bf16(v16bf a, v16bf b, v8f c) {
  // D = A(16x32 bf16) * B(32x16 bf16) + C(16x16 f32)
  return __builtin_amdgcn_wmma_f32_16x16x32_bf16(
      /*neg_a=*/false, a, /*neg_b=*/false, b,
      /*c_mod=*/(short)0, c, /*reuse_a=*/false, /*reuse_b=*/false);
}

// A-operand (16x32, MxK): lane l<16 holds row l, K in {k0..k0+8} u {k0+16..k0+24};
// lane l+16 holds row l, K shifted by +8 in each half. (ISA 7.12.2, 16-bit A)
static __device__ __forceinline__ v16bf load_a16(const __bf16* base, int row,
                                                 int k0, int hf) {
  union { v16bf v; v8bf h[2]; } u;
  const __bf16* p = base + row * DIM;
  u.h[0] = *(const v8bf*)(p + k0 + hf * 8);
  u.h[1] = *(const v8bf*)(p + k0 + 16 + hf * 8);
  return u.v;
}

// B-operand (32x16, KxN): lane n<16 holds col n, K in {k0..k0+8} u {k0+16..k0+24};
// lane n+16 holds col n, K shifted by +8 in each half (matches A).
static __device__ __forceinline__ v16bf load_b16(const __bf16* base, int colLocal,
                                                 int k0, int hf) {
  union { v16bf v; v8bf h[2]; } u;
  const __bf16* p = base + colLocal * DIM + k0 + hf * 8;
  u.h[0] = *(const v8bf*)(p);
  u.h[1] = *(const v8bf*)(p + 16);
  return u.v;
}

// ---- gfx1250 async memory->LDS copy (ASYNCcnt-tracked, no VGPR round trip) ----
// GVS addressing: mem = SGPR base + VGPR byte offset; LDS dest = low 32 bits of
// the generic shared pointer (generic LDS addr = {SHARED_BASE, lds_offset}).
static __device__ __forceinline__ void async_b128(const void* lds_dst,
                                                  const void* gbase, int off) {
  unsigned d = (unsigned)(unsigned long long)lds_dst;
  asm volatile("global_load_async_to_lds_b128 %0, %1, %2"
               :: "v"(d), "v"(off), "s"(gbase)
               : "memory");
}
static __device__ __forceinline__ void wait_async0() {
  asm volatile("s_wait_asynccnt 0" ::: "memory");
}

// Stage one 64-code chunk (32KB hi + 32KB lo) via async DMA; 16 ops/thread.
static __device__ __forceinline__ void stage_b_async(const char* lds_base,  // smem
                                                     int buf_off,
                                                     const __bf16* e_hi,
                                                     const __bf16* e_lo,
                                                     int cb, int tid) {
  const char* ghi = (const char*)(e_hi + (long)cb * DIM);
  const char* glo = (const char*)(e_lo + (long)cb * DIM);
#pragma unroll
  for (int i = 0; i < 8; ++i) {
    const int off = (tid + i * 256) * 16;  // covers 32768 bytes
    async_b128(lds_base + buf_off + off, ghi, off);
    async_b128(lds_base + buf_off + LDS_BLO + off, glo, off);
  }
}

// -------- Kernel 1: split codebook into bf16 hi/lo + ||e||^2 (f32) --------
__global__ void __launch_bounds__(256)
vq_prep(const float* __restrict__ emb,
        __bf16* __restrict__ e_hi, __bf16* __restrict__ e_lo,
        float* __restrict__ ebsq) {
  const int r = blockIdx.x, t = threadIdx.x;
  const float x = emb[r * DIM + t];
  const __bf16 hi = (__bf16)x;
  e_hi[r * DIM + t] = hi;
  e_lo[r * DIM + t] = (__bf16)(x - (float)hi);
  __shared__ float red[256];
  red[t] = x * x;
  __syncthreads();
  for (int s = 128; s > 0; s >>= 1) {
    if (t < s) red[t] += red[t + s];
    __syncthreads();
  }
  if (t == 0) ebsq[r] = red[0];
}

// -------- Kernel 2: WMMA distance GEMM + fused argmin (double-buffered async B) ----
__global__ void __launch_bounds__(256)
vq_argmin(const float* __restrict__ z,
          const __bf16* __restrict__ e_hi, const __bf16* __restrict__ e_lo,
          const float* __restrict__ ebsq,
          float* __restrict__ out, int* __restrict__ idx_ws) {
  extern __shared__ __align__(16) char smem[];
  __bf16* a_hi = (__bf16*)(smem + LDS_AHI);   // 128x256 bf16 = 64KB
  __bf16* a_lo = (__bf16*)(smem + LDS_ALO);   // 64KB

  const int tid = threadIdx.x;
  const int lane = tid & 31;
  const int w = tid >> 5;          // wave 0..7
  const int hf = lane >> 4;        // 0: lanes 0-15, 1: lanes 16-31
  const int ln = lane & 15;
  const int rowbase = w * 16;      // 16 token rows per wave
  const long tokBase = (long)blockIdx.x * TOK_PER_BLK;

  // Kick off the async DMA of codebook chunk 0 before touching A.
  stage_b_async(smem, LDS_B, e_hi, e_lo, 0, tid);

  // Stage A: z block -> bf16 hi/lo in LDS (split compensates bf16 rounding)
  const float4* zb = (const float4*)(z + tokBase * DIM);
  for (int i = tid; i < TOK_PER_BLK * DIM / 4; i += 256) {
    const float4 v = zb[i];
    const float vv[4] = {v.x, v.y, v.z, v.w};
#pragma unroll
    for (int c2 = 0; c2 < 4; ++c2) {
      const float x = vv[c2];
      const __bf16 hi = (__bf16)x;
      a_hi[4 * i + c2] = hi;
      a_lo[4 * i + c2] = (__bf16)(x - (float)hi);
    }
  }
  wait_async0();       // my chunk-0 DMA complete
  __syncthreads();     // everyone's A stores + chunk-0 DMA visible

  float best[8];
  int bidx[8];
#pragma unroll
  for (int j = 0; j < 8; ++j) { best[j] = 3.4e38f; bidx[j] = 0; }

  const v8f vzero = {0.f, 0.f, 0.f, 0.f, 0.f, 0.f, 0.f, 0.f};

  for (int c = 0; c < NCHUNK; ++c) {
    const int cb = c * CODES_PER_CHUNK;
    // Integer-offset buffer select keeps one GEP chain off `smem`
    // => InferAddressSpaces keeps these as AS(3) => ds_load_b128.
    const int boff = LDS_B + (c & 1) * LDS_BSTEP;
    const int noff = LDS_B + (1 - (c & 1)) * LDS_BSTEP;

    // Prefetch next chunk into the other buffer; overlaps with the WMMAs below.
    if (c + 1 < NCHUNK)
      stage_b_async(smem, noff, e_hi, e_lo, cb + CODES_PER_CHUNK, tid);

    const __bf16* bh = (const __bf16*)(smem + boff);
    const __bf16* bl = (const __bf16*)(smem + boff + LDS_BLO);

    v8f acc[4];
#pragma unroll
    for (int t = 0; t < 4; ++t) acc[t] = vzero;

#pragma unroll
    for (int kc = 0; kc < 8; ++kc) {
      const int k0 = kc * 32;
      const v16bf ahi = load_a16(a_hi, rowbase + ln, k0, hf);
      const v16bf alo = load_a16(a_lo, rowbase + ln, k0, hf);
#pragma unroll
      for (int t = 0; t < 4; ++t) {
        const v16bf bhi = load_b16(bh, t * 16 + ln, k0, hf);
        const v16bf blo = load_b16(bl, t * 16 + ln, k0, hf);
        // dot ~= zlo*ehi + zhi*elo + zhi*ehi  (f32 accumulate)
        acc[t] = wmma_bf16(alo, bhi, acc[t]);
        acc[t] = wmma_bf16(ahi, blo, acc[t]);
        acc[t] = wmma_bf16(ahi, bhi, acc[t]);
      }
    }

    // Fused argmin epilogue on C/D layout: lane owns col ln, rows j+8*hf
#pragma unroll
    for (int t = 0; t < 4; ++t) {
      const int col = cb + t * 16 + ln;
      const float eb = ebsq[col];
#pragma unroll
      for (int j = 0; j < 8; ++j) {
        const float s = eb - 2.0f * acc[t][j];
        if (s < best[j]) { best[j] = s; bidx[j] = col; }
      }
    }

    wait_async0();     // my next-chunk DMA landed
    __syncthreads();   // all waves done reading cur + all DMAs visible
  }

  // Cross-lane argmin over the 16 columns per half-wave (wave32 shfl)
#pragma unroll
  for (int j = 0; j < 8; ++j) {
    float bv = best[j];
    int bi = bidx[j];
#pragma unroll
    for (int m = 1; m < 16; m <<= 1) {
      const float ov = __shfl_xor(bv, m, 32);
      const int oi = __shfl_xor(bi, m, 32);
      if (ov < bv || (ov == bv && oi < bi)) { bv = ov; bi = oi; }  // first-index tiebreak
    }
    if (ln == 0) {
      const long row = tokBase + rowbase + j + 8 * hf;
      out[OUT_IDX + row] = (float)bi;
      idx_ws[row] = bi;
    }
  }
}

// -------- Kernel 3: gather z_q, write straight-through output, loss partials --------
__global__ void __launch_bounds__(256)
vq_gather(const float* __restrict__ z, const float* __restrict__ emb,
          const int* __restrict__ idx_ws,
          float* __restrict__ out, float* __restrict__ partials) {
  const int tid = threadIdx.x;
  const long g = (long)blockIdx.x * 256 + tid;
  const long token = g >> 6;
  const int j = (int)(g & 63);
  const int idx = idx_ws[token];
  const float4 e4 = ((const float4*)emb)[(long)idx * 64 + j];
  const float4 z4 = ((const float4*)z)[token * 64 + j];
  float4 o;  // z + (z_q - z), matching the reference's fp ordering
  o.x = z4.x + (e4.x - z4.x);
  o.y = z4.y + (e4.y - z4.y);
  o.z = z4.z + (e4.z - z4.z);
  o.w = z4.w + (e4.w - z4.w);
  ((float4*)out)[token * 64 + j] = o;
  const float dx = z4.x - e4.x, dy = z4.y - e4.y, dz = z4.z - e4.z, dw = z4.w - e4.w;
  __shared__ float red[256];
  red[tid] = dx * dx + dy * dy + dz * dz + dw * dw;
  __syncthreads();
  for (int s = 128; s > 0; s >>= 1) {
    if (tid < s) red[tid] += red[tid + s];
    __syncthreads();
  }
  if (tid == 0) partials[blockIdx.x] = red[0];
}

// -------- Kernel 4: deterministic final reduction for the loss --------
__global__ void __launch_bounds__(256)
vq_finalize(const float* __restrict__ partials, float* __restrict__ out) {
  __shared__ double red[256];
  const int tid = threadIdx.x;
  double s = 0.0;
  for (int i = tid; i < GATHER_BLOCKS; i += 256) s += (double)partials[i];
  red[tid] = s;
  __syncthreads();
  for (int k = 128; k > 0; k >>= 1) {
    if (tid < k) red[tid] += red[tid + k];
    __syncthreads();
  }
  if (tid == 0)
    out[OUT_LOSS] = (float)(1.25 * red[0] / (double)((long)N_TOKENS * DIM));
}

extern "C" void kernel_launch(void* const* d_in, const int* in_sizes, int n_in,
                              void* d_out, int out_size, void* d_ws, size_t ws_size,
                              hipStream_t stream) {
  const float* z = (const float*)d_in[0];     // [65536, 256] f32
  const float* emb = (const float*)d_in[1];   // [1024, 256]  f32
  float* out = (float*)d_out;
  char* ws = (char*)d_ws;

  // Workspace layout (~1.38 MB)
  __bf16* e_hi = (__bf16*)(ws);                    // 512 KB
  __bf16* e_lo = (__bf16*)(ws + 524288);           // 512 KB
  float* ebsq = (float*)(ws + 1048576);            // 4 KB
  int* idxw = (int*)(ws + 1052672);                // 256 KB
  float* part = (float*)(ws + 1314816);            // 64 KB

  vq_prep<<<K_CODES, 256, 0, stream>>>(emb, e_hi, e_lo, ebsq);
  vq_argmin<<<N_TOKENS / TOK_PER_BLK, 256, 262144, stream>>>(z, e_hi, e_lo, ebsq,
                                                             out, idxw);
  vq_gather<<<GATHER_BLOCKS, 256, 0, stream>>>(z, emb, idxw, out, part);
  vq_finalize<<<1, 256, 0, stream>>>(part, out);
}
